// SparseFlexAttn_5574867550977
// MI455X (gfx1250) — matrix operand order repaired and checked by reading, not compiled, not run on hardware
//
#include <hip/hip_runtime.h>
#include <hip/hip_bf16.h>

typedef __attribute__((ext_vector_type(16))) _Float16 v16h;
typedef __attribute__((ext_vector_type(8)))  _Float16 h8;
typedef __attribute__((ext_vector_type(4)))  _Float16 h4;
typedef __attribute__((ext_vector_type(2)))  _Float16 h2;
typedef __attribute__((ext_vector_type(8)))  float    v8f;

#define HEADS 16
#define SQ    1600
#define SKV   4121
#define DH    128
#define NQB   13
#define NKB   33
#define LSTR  136                       // LDS row stride in halves (272B = 68 dwords -> bank 4n, conflict free)
#define QK_SCALE 0.08838834764831845f   // 1/sqrt(128)

// ---------------------------------------------------------------------------
// Packed f32->f16 conversion helpers (v_cvt_pk_rtz_f16_f32: 2 elems/instr).
// ---------------------------------------------------------------------------
__device__ __forceinline__ h2 pk2(float a, float b) {
    return (h2)__builtin_amdgcn_cvt_pkrtz(a, b);
}
__device__ __forceinline__ h4 pk4(float a, float b, float c, float d) {
    h2 lo = pk2(a, b), hi = pk2(c, d);
    return __builtin_shufflevector(lo, hi, 0, 1, 2, 3);
}
__device__ __forceinline__ h8 pk8(float4 a, float4 b) {
    h4 lo = pk4(a.x, a.y, a.z, a.w), hi = pk4(b.x, b.y, b.z, b.w);
    return __builtin_shufflevector(lo, hi, 0, 1, 2, 3, 4, 5, 6, 7);
}

// ---------------------------------------------------------------------------
// Setup kernel: rebuild the reference's 128x128 any-reduced block mask in d_ws.
// ---------------------------------------------------------------------------
__global__ void SparseFlexAttn_mask_kernel(int* __restrict__ mask) {
    const int kb = blockIdx.x;  // 0..32
    const int qb = blockIdx.y;  // 0..12
    __shared__ int flag;
    if (threadIdx.x == 0) flag = 0;
    __syncthreads();

    const int slice[12] = {0, 1, 5, 21, 57, 121, 265, 521, 921, 1497, 2521, 4121};
    const int WK[11]    = {1, 2, 4, 6, 8, 12, 16, 20, 24, 32, 40};
    const int HK[11]    = {1, 1, 2, 2, 3, 3, 4, 4, 5, 5, 6};  // kernel // 2

    bool any = false;
    for (int idx = threadIdx.x; idx < 128 * 128; idx += blockDim.x) {
        const int qi = qb * 128 + (idx >> 7);
        const int kj = kb * 128 + (idx & 127);
        if (qi >= SQ || kj >= SKV) continue;
        if (kj < 121) { any = true; continue; }     // attention sink (scales 0..4)
        int s = 0;
        #pragma unroll
        for (int j = 1; j < 11; ++j) if (kj >= slice[j]) s = j;
        const int rel = kj - slice[s];
        const int Wk  = WK[s];
        const int ky  = rel / Wk;
        const int kx  = rel - ky * Wk;
        const int qy  = qi / 40;
        const int qx  = qi - qy * 40;
        // np.round == rintf (round half to even), computed in f32 like the reference
        const float cy = rintf((float)qy / 40.0f * (float)Wk);
        const float cx = rintf((float)qx / 40.0f * (float)Wk);
        const float hk = (float)HK[s];
        if (fabsf(cy - (float)ky) <= hk && fabsf(cx - (float)kx) <= hk) any = true;
    }
    if (any) flag = 1;
    __syncthreads();
    if (threadIdx.x == 0) mask[qb * NKB + kb] = flag;
}

// ---------------------------------------------------------------------------
// LDS operand gathers (16B aligned b128 pairs).
// ---------------------------------------------------------------------------
__device__ __forceinline__ v16h lds_pair(const _Float16* p0, const _Float16* p1) {
    h8 lo = *(const h8*)p0;
    h8 hi = *(const h8*)p1;
    return __builtin_shufflevector(lo, hi, 0, 1, 2, 3, 4, 5, 6, 7,
                                           8, 9, 10, 11, 12, 13, 14, 15);
}
// B operand (32x16): lane n -> row (16t+n) of the staged matrix, halves {32c+8lh+0..7, 32c+16+8lh+0..7}.
__device__ __forceinline__ v16h lds_B(const _Float16* base, int t, int c, int l15, int lh) {
    const _Float16* p = base + (16 * t + l15) * LSTR + 32 * c + 8 * lh;
    return lds_pair(p, p + 16);
}
// A operand (16x32): lane -> row (l15), halves {32c+8lh+0..7, 32c+16+8lh+0..7}.
__device__ __forceinline__ v16h lds_A(const _Float16* base, int c, int l15, int lh) {
    const _Float16* p = base + l15 * LSTR + 32 * c + 8 * lh;
    return lds_pair(p, p + 16);
}

__device__ __forceinline__ float redmax16(float v) {
    #pragma unroll
    for (int m = 1; m < 16; m <<= 1) v = fmaxf(v, __shfl_xor(v, m, 32));
    return v;
}
__device__ __forceinline__ float redsum16(float v) {
    #pragma unroll
    for (int m = 1; m < 16; m <<= 1) v += __shfl_xor(v, m, 32);
    return v;
}

// ---------------------------------------------------------------------------
// Fused block-sparse flash attention. grid = (16 heads, 13 q-blocks), 256 thr.
// ---------------------------------------------------------------------------
__launch_bounds__(256, 1)
__global__ void SparseFlexAttn_5574867550977_kernel(
        const float* __restrict__ Qg, const float* __restrict__ Kg,
        const float* __restrict__ Vg, float* __restrict__ Og,
        const int* __restrict__ mask) {
    extern __shared__ _Float16 smem[];
    _Float16* Ks = smem;                   // 128 x LSTR  (K block, row-major over kv)
    _Float16* Vt = smem + 128 * LSTR;      // 128 x LSTR  (V block, transposed: [d][kv])
    _Float16* Ps = smem + 256 * LSTR;      // 8 waves x 16 rows x LSTR (P strips)

    const int h    = blockIdx.x;
    const int qb   = blockIdx.y;
    const int tid  = threadIdx.x;
    const int lane = tid & 31;
    const int wv   = tid >> 5;       // wave 0..7 -> q rows [qb*128 + wv*16, +16)
    const int l15  = lane & 15;
    const int lh   = lane >> 4;

    // ---- Q strip -> registers in WMMA A layout, pre-scaled, f16 ----
    const bool qvalid = (qb * 128 + wv * 16) < SQ;
    const int  qr     = min(qb * 128 + wv * 16 + l15, SQ - 1);
    const float* qp   = Qg + ((size_t)h * SQ + qr) * DH;
    v16h Qa[4];
    #pragma unroll
    for (int c = 0; c < 4; ++c) {
        const int d0 = 32 * c + 8 * lh;
        float4 a0 = *(const float4*)(qp + d0);
        float4 a1 = *(const float4*)(qp + d0 + 4);
        float4 b0 = *(const float4*)(qp + d0 + 16);
        float4 b1 = *(const float4*)(qp + d0 + 20);
        a0.x *= QK_SCALE; a0.y *= QK_SCALE; a0.z *= QK_SCALE; a0.w *= QK_SCALE;
        a1.x *= QK_SCALE; a1.y *= QK_SCALE; a1.z *= QK_SCALE; a1.w *= QK_SCALE;
        b0.x *= QK_SCALE; b0.y *= QK_SCALE; b0.z *= QK_SCALE; b0.w *= QK_SCALE;
        b1.x *= QK_SCALE; b1.y *= QK_SCALE; b1.z *= QK_SCALE; b1.w *= QK_SCALE;
        h8 lo = pk8(a0, a1);
        h8 hi = pk8(b0, b1);
        Qa[c] = __builtin_shufflevector(lo, hi, 0, 1, 2, 3, 4, 5, 6, 7,
                                                8, 9, 10, 11, 12, 13, 14, 15);
    }

    v8f O[8];
    #pragma unroll
    for (int t = 0; t < 8; ++t) { v8f z = {0, 0, 0, 0, 0, 0, 0, 0}; O[t] = z; }
    float mrow[8], lrow[8];
    #pragma unroll
    for (int p = 0; p < 8; ++p) { mrow[p] = -__builtin_inff(); lrow[p] = 0.0f; }

    for (int kb = 0; kb < NKB; ++kb) {
        if (!mask[qb * NKB + kb]) continue;     // uniform (scalar) block skip
        const int kv0 = kb * 128;
        __syncthreads();                         // previous iteration's LDS reads done

        // ---- stage K block: 128 kv rows x 128 d, f32 -> f16 (pkrtz) ----
        {
            const int row = tid >> 1;
            const int cb  = (tid & 1) * 64;
            const int kvr = min(kv0 + row, SKV - 1);
            const float* kp = Kg + ((size_t)h * SKV + kvr) * DH + cb;
            _Float16* dst = Ks + row * LSTR + cb;
            #pragma unroll
            for (int j = 0; j < 64; j += 8) {
                float4 a = *(const float4*)(kp + j);
                float4 b = *(const float4*)(kp + j + 4);
                *(h8*)(dst + j) = pk8(a, b);
            }
        }
        // ---- stage V block transposed: Vt[d][kv], b64 stores along kv ----
        {
            const int dg = (tid & 7) * 16;   // d group of 16
            const int rg = (tid >> 3) * 4;   // 4 consecutive kv rows
            const float* v0 = Vg + ((size_t)h * SKV + min(kv0 + rg + 0, SKV - 1)) * DH + dg;
            const float* v1 = Vg + ((size_t)h * SKV + min(kv0 + rg + 1, SKV - 1)) * DH + dg;
            const float* v2 = Vg + ((size_t)h * SKV + min(kv0 + rg + 2, SKV - 1)) * DH + dg;
            const float* v3 = Vg + ((size_t)h * SKV + min(kv0 + rg + 3, SKV - 1)) * DH + dg;
            #pragma unroll
            for (int j = 0; j < 16; j += 4) {
                float4 a = *(const float4*)(v0 + j);
                float4 b = *(const float4*)(v1 + j);
                float4 c = *(const float4*)(v2 + j);
                float4 d = *(const float4*)(v3 + j);
                *(h4*)(Vt + (dg + j + 0) * LSTR + rg) = pk4(a.x, b.x, c.x, d.x);
                *(h4*)(Vt + (dg + j + 1) * LSTR + rg) = pk4(a.y, b.y, c.y, d.y);
                *(h4*)(Vt + (dg + j + 2) * LSTR + rg) = pk4(a.z, b.z, c.z, d.z);
                *(h4*)(Vt + (dg + j + 3) * LSTR + rg) = pk4(a.w, b.w, c.w, d.w);
            }
        }
        __syncthreads();

        // ---- S = (Q*scale) @ K^T : 8 col-tiles x 4 K-chunks of WMMA ----
        v8f S[8];
        #pragma unroll
        for (int t = 0; t < 8; ++t) {
            v8f acc = {0, 0, 0, 0, 0, 0, 0, 0};
            #pragma unroll
            for (int c = 0; c < 4; ++c) {
                v16h b = lds_B(Ks, t, c, l15, lh);
                acc = __builtin_amdgcn_wmma_f32_16x16x32_f16(
                          false, Qa[c], false, b, (short)0, acc, false, false);
            }
            S[t] = acc;
        }
        // mask padded kv columns of the last block
        if (kv0 + 128 > SKV) {
            #pragma unroll
            for (int t = 0; t < 8; ++t) {
                const bool bad = (kv0 + 16 * t + l15) >= SKV;
                #pragma unroll
                for (int p = 0; p < 8; ++p) S[t][p] = bad ? -1.0e30f : S[t][p];
            }
        }

        // ---- online softmax (rows live in VGPR index p; cols across 16 lanes) ----
        float alpha[8];
        #pragma unroll
        for (int p = 0; p < 8; ++p) {
            float mx = S[0][p];
            #pragma unroll
            for (int t = 1; t < 8; ++t) mx = fmaxf(mx, S[t][p]);
            mx = redmax16(mx);
            const float mn = fmaxf(mrow[p], mx);
            alpha[p] = __expf(mrow[p] - mn);
            mrow[p]  = mn;
        }
        float ls[8] = {0, 0, 0, 0, 0, 0, 0, 0};
        #pragma unroll
        for (int t = 0; t < 8; ++t) {
            #pragma unroll
            for (int p = 0; p < 8; ++p) {
                const float e = __expf(S[t][p] - mrow[p]);
                S[t][p] = e;
                ls[p]  += e;
            }
        }
        #pragma unroll
        for (int p = 0; p < 8; ++p) {
            lrow[p] = lrow[p] * alpha[p] + redsum16(ls[p]);
            #pragma unroll
            for (int t = 0; t < 8; ++t) O[t][p] *= alpha[p];
        }

        // ---- P -> per-wave LDS strip (f16), then O += P @ V ----
        _Float16* ps = Ps + wv * 16 * LSTR;
        #pragma unroll
        for (int t = 0; t < 8; ++t) {
            #pragma unroll
            for (int p = 0; p < 8; ++p)
                ps[(8 * lh + p) * LSTR + 16 * t + l15] = (_Float16)S[t][p];
        }
        asm volatile("s_wait_dscnt 0x0" ::: "memory");  // same-wave LDS RAW
        #pragma unroll
        for (int c = 0; c < 4; ++c) {
            v16h A = lds_A(ps, c, l15, lh);
            #pragma unroll
            for (int t = 0; t < 8; ++t) {
                v16h Bv = lds_B(Vt, t, c, l15, lh);
                O[t] = __builtin_amdgcn_wmma_f32_16x16x32_f16(
                           false, A, false, Bv, (short)0, O[t], false, false);
            }
        }
    }

    // ---- normalize and store ----
    if (qvalid) {
        #pragma unroll
        for (int p = 0; p < 8; ++p) {
            const float inv = 1.0f / lrow[p];
            const int row = qb * 128 + wv * 16 + 8 * lh + p;
            float* op = Og + ((size_t)h * SQ + row) * DH;
            #pragma unroll
            for (int t = 0; t < 8; ++t) op[16 * t + l15] = O[t][p] * inv;
        }
    }
}

extern "C" void kernel_launch(void* const* d_in, const int* in_sizes, int n_in,
                              void* d_out, int out_size, void* d_ws, size_t ws_size,
                              hipStream_t stream) {
    (void)in_sizes; (void)n_in; (void)out_size; (void)ws_size;
    const float* q = (const float*)d_in[0];
    const float* k = (const float*)d_in[1];
    const float* v = (const float*)d_in[2];
    float* out = (float*)d_out;
    int* mask = (int*)d_ws;   // 13*33 ints

    SparseFlexAttn_mask_kernel<<<dim3(NKB, NQB), 256, 0, stream>>>(mask);

    const size_t smem_bytes = (size_t)3 * 128 * LSTR * sizeof(_Float16);  // ~104 KB of 320 KB WGP LDS
    SparseFlexAttn_5574867550977_kernel<<<dim3(HEADS, NQB), 256, smem_bytes, stream>>>(
        q, k, v, out, mask);
}
